// BahdanauAttention_13623636263699
// MI455X (gfx1250) — matrix operand order repaired and checked by reading, not compiled, not run on hardware
//
#include <hip/hip_runtime.h>

// Problem sizes (fixed by the reference).
#define BB 32
#define SS 2048
#define EE 1024
#define DD 1024

typedef __attribute__((ext_vector_type(16))) __bf16 v16bf;
typedef __attribute__((ext_vector_type(8)))  float  v8f;
typedef __attribute__((ext_vector_type(4)))  float  v4f;

// ---------------------------------------------------------------------------
// Kernel 1: convert We (f32 [D,E]) -> bf16, once. We is 4 MB -> L2 resident.
// ---------------------------------------------------------------------------
__global__ __launch_bounds__(256) void bah_convert_we(const float* __restrict__ We,
                                                      __bf16* __restrict__ WeBf) {
    int i = (blockIdx.x * 256 + threadIdx.x) * 4;
    v4f x = *(const v4f*)(We + i);
    WeBf[i + 0] = (__bf16)x.x;
    WeBf[i + 1] = (__bf16)x.y;
    WeBf[i + 2] = (__bf16)x.z;
    WeBf[i + 3] = (__bf16)x.w;
}

// ---------------------------------------------------------------------------
// Kernel 2: dec_proj[b,d] = sum_k hidden[b,k] * Wd[d,k]   (32x1024x1024 GEMM)
// One wave per block; grid = 2 M-tiles x 64 N-tiles. bf16 WMMA, f32 accum.
// A (16-bit 16x32) lane layout used here: lane L holds row M=L%16, K = (L/16)*16+{0..7}
// and (L/16)*16+8+{0..7}.  B (32x16) lane layout: lane L holds col N=L%16,
// K = (L/16)*16 + {0..15} (contiguous along K = row-major Wd row).
// ---------------------------------------------------------------------------
__global__ __launch_bounds__(32) void bah_dec_proj(const float* __restrict__ hidden,
                                                   const float* __restrict__ Wd,
                                                   float* __restrict__ dec) {
    const int mtile = blockIdx.x & 1;        // 2 tiles of 16 b-rows
    const int ntile = blockIdx.x >> 1;       // 64 tiles of 16 d-cols
    const int lane  = threadIdx.x;
    const int nloc  = lane & 15;
    const int khalf = lane >> 4;

    v8f acc = {};
    for (int k0 = 0; k0 < DD; k0 += 32) {
        v16bf af, bf;
        const float* h = hidden + (size_t)(mtile * 16 + nloc) * DD;
#pragma unroll
        for (int j = 0; j < 8; ++j) {
            af[j]     = (__bf16)h[k0 + khalf * 16 + j];
            af[8 + j] = (__bf16)h[k0 + khalf * 16 + 8 + j];
        }
        const float* wp = Wd + (size_t)(ntile * 16 + nloc) * DD + k0 + khalf * 16;
#pragma unroll
        for (int j = 0; j < 16; ++j) bf[j] = (__bf16)wp[j];

        acc = __builtin_amdgcn_wmma_f32_16x16x32_bf16(false, af, false, bf,
                                                      (short)0, acc, false, false);
    }
    // C layout: VGPR r, lane L -> row M = r + 8*(L/16), col N = L%16.
#pragma unroll
    for (int r = 0; r < 8; ++r) {
        int brow = mtile * 16 + r + (khalf << 3);
        int d    = ntile * 16 + nloc;
        dec[brow * DD + d] = acc[r];
    }
}

// ---------------------------------------------------------------------------
// Kernel 3 (main): fused  scores[b,s] = sum_d v[d]*tanh(enc_proj[b,s,d]+dec[b,d])
// Block = 256 threads (8 waves), handles one batch b and 64 S-rows.
//   wave w owns N-group w (128 d-cols = 8 N-tiles) and ALL 4 M-tiles, so each
//   B fragment feeds 4 WMMAs and no two waves fetch the same B data.
//   We traffic: 1024 blocks x 2 MB = 2 GB of L2 reads.
// K loop over E in steps of 32, software-pipelined: the NEXT k-step's A values
// are loaded into registers right after the first barrier, so their
// s_wait_loadcnt lands after the whole WMMA block. B fragments are prefetched
// one k-step ahead (global_prefetch_b8, speculative -> OOB-safe).
// A tile staged f32->bf16 into LDS pre-swizzled to the WMMA A lane layout so
// each lane's fragment is one contiguous 32B ds_load.
// Epilogue folds dec_proj + tanh + v-dot, reduces across lanes (shfl_xor) and
// across the 8 N-group waves (LDS ds_add_f32).
// ---------------------------------------------------------------------------
__global__ __launch_bounds__(256) void bah_scores(const float* __restrict__ enc,
                                                  const __bf16* __restrict__ WeBf,
                                                  const float* __restrict__ dec,
                                                  const float* __restrict__ vvec,
                                                  float* __restrict__ scores) {
    __shared__ __align__(32) __bf16 Asw[2048];   // 4 M-tiles x 32 lanes x 16 bf16
    __shared__ float s_scores[64];

    const int tid  = threadIdx.x;
    const int b    = blockIdx.x >> 5;            // 32 S-tiles per batch
    const int s0   = (blockIdx.x & 31) * 64;
    const int lane = tid & 31;
    const int dbase0 = (tid >> 5) * 128;         // wave's N-group (8 tiles)
    const int nloc   = lane & 15;
    const int khalf  = lane >> 4;

    if (tid < 64) s_scores[tid] = 0.0f;

    v8f acc[4][8] = {};                          // [M-tile][N-tile] accumulators

    // Per-thread staging addresses (8 elements of the 64x32 A tile each).
    int   sl8[8], kk8[8];
#pragma unroll
    for (int t = 0; t < 8; ++t) {
        int idx = tid + t * 256;                 // 0..2047
        sl8[t] = idx >> 5;                       // local s row (0..63)
        kk8[t] = idx & 31;                       // local k
    }

    // ---- prologue: load k0 = 0 tile into registers ----
    float areg[8];
#pragma unroll
    for (int t = 0; t < 8; ++t)
        areg[t] = enc[((size_t)(s0 + sl8[t]) * BB + b) * EE + kk8[t]];

    for (int k0 = 0; k0 < EE; k0 += 32) {
        // ---- commit staged registers to swizzled bf16 LDS ----
#pragma unroll
        for (int t = 0; t < 8; ++t) {
            int sl = sl8[t], k = kk8[t];
            int tile = sl >> 4;
            int M    = sl & 15;
            int half = k >> 4;                   // which 16-lane half holds it
            int elem = k & 15;
            Asw[tile * 512 + (M + (half << 4)) * 16 + elem] = (__bf16)areg[t];
        }
        __syncthreads();

        // ---- issue next k-step's A loads early (latency hidden by WMMAs) ----
        if (k0 + 32 < EE) {
#pragma unroll
            for (int t = 0; t < 8; ++t)
                areg[t] = enc[((size_t)(s0 + sl8[t]) * BB + b) * EE
                              + (k0 + 32) + kk8[t]];
        }

        v16bf af[4];
#pragma unroll
        for (int m = 0; m < 4; ++m)
            af[m] = *(const v16bf*)(Asw + m * 512 + lane * 16);

#pragma unroll
        for (int nt = 0; nt < 8; ++nt) {
            const __bf16* bp = WeBf + (size_t)(dbase0 + nt * 16 + nloc) * EE
                                    + k0 + (khalf << 4);
            __builtin_prefetch(bp + 32, 0, 0);   // next k-step's fragment
            v16bf bfr = *(const v16bf*)bp;
#pragma unroll
            for (int m = 0; m < 4; ++m)
                acc[m][nt] = __builtin_amdgcn_wmma_f32_16x16x32_bf16(
                    false, af[m], false, bfr, (short)0, acc[m][nt], false, false);
        }
        __syncthreads();
    }

    // ---- fused epilogue: + dec_proj, tanh, * v, row reduction ----
    float part[4][8];
#pragma unroll
    for (int m = 0; m < 4; ++m)
#pragma unroll
        for (int r = 0; r < 8; ++r) part[m][r] = 0.0f;

#pragma unroll
    for (int nt = 0; nt < 8; ++nt) {
        int d    = dbase0 + nt * 16 + nloc;
        float dv = dec[b * DD + d];
        float vv = vvec[d];
#pragma unroll
        for (int m = 0; m < 4; ++m)
#pragma unroll
            for (int r = 0; r < 8; ++r)
                part[m][r] += vv * tanhf(acc[m][nt][r] + dv);
    }

    // reduce over the 16 N-lanes (xor masks 8,4,2,1 stay inside each half)
#pragma unroll
    for (int m = 0; m < 4; ++m)
#pragma unroll
        for (int r = 0; r < 8; ++r) {
#pragma unroll
            for (int x = 8; x >= 1; x >>= 1)
                part[m][r] += __shfl_xor(part[m][r], x, 32);
        }

    if (nloc == 0) {
#pragma unroll
        for (int m = 0; m < 4; ++m)
#pragma unroll
            for (int r = 0; r < 8; ++r)
                atomicAdd(&s_scores[m * 16 + r + (khalf << 3)], part[m][r]);
    }
    __syncthreads();
    if (tid < 64) scores[b * SS + s0 + tid] = s_scores[tid];
}

// ---------------------------------------------------------------------------
// Kernel 4: softmax over S per batch row -> attention weights (into d_out).
// ---------------------------------------------------------------------------
__global__ __launch_bounds__(256) void bah_softmax(const float* __restrict__ scores,
                                                   float* __restrict__ wts) {
    __shared__ float red[256];
    const int b = blockIdx.x, tid = threadIdx.x;

    float x[8];
    float mx = -3.4e38f;
#pragma unroll
    for (int i = 0; i < 8; ++i) {
        x[i] = scores[b * SS + tid + i * 256];
        mx = fmaxf(mx, x[i]);
    }
    red[tid] = mx;
    __syncthreads();
    for (int s = 128; s > 0; s >>= 1) {
        if (tid < s) red[tid] = fmaxf(red[tid], red[tid + s]);
        __syncthreads();
    }
    mx = red[0];
    __syncthreads();

    float sum = 0.0f;
#pragma unroll
    for (int i = 0; i < 8; ++i) {
        x[i] = expf(x[i] - mx);
        sum += x[i];
    }
    red[tid] = sum;
    __syncthreads();
    for (int s = 128; s > 0; s >>= 1) {
        if (tid < s) red[tid] += red[tid + s];
        __syncthreads();
    }
    float inv = 1.0f / red[0];
#pragma unroll
    for (int i = 0; i < 8; ++i)
        wts[b * SS + tid + i * 256] = x[i] * inv;
}

// ---------------------------------------------------------------------------
// Kernel 5: context partials. block = (b, s-chunk of 128); threads cover E via
// float4. Deterministic (fixed-order accumulation, no FP global atomics).
// ---------------------------------------------------------------------------
__global__ __launch_bounds__(256) void bah_ctx_partial(const float* __restrict__ enc,
                                                       const float* __restrict__ wts,
                                                       float* __restrict__ parts) {
    const int b = blockIdx.x >> 4;
    const int c = blockIdx.x & 15;
    const int e = threadIdx.x * 4;
    v4f acc = {};
    const int sbase = c * 128;
    for (int i = 0; i < 128; ++i) {
        int s = sbase + i;
        float wv = wts[b * SS + s];
        v4f f = *(const v4f*)(enc + ((size_t)s * BB + b) * EE + e);
        acc.x += wv * f.x;
        acc.y += wv * f.y;
        acc.z += wv * f.z;
        acc.w += wv * f.w;
    }
    *(v4f*)(parts + ((size_t)c * BB + b) * EE + e) = acc;
}

// Kernel 6: reduce the 16 s-chunk partials -> context (into d_out).
__global__ __launch_bounds__(256) void bah_ctx_reduce(const float* __restrict__ parts,
                                                      float* __restrict__ ctx) {
    int gid = blockIdx.x * 256 + threadIdx.x;   // b*EE + e
    float s = 0.0f;
#pragma unroll
    for (int c = 0; c < 16; ++c)
        s += parts[(size_t)c * BB * EE + gid];
    ctx[gid] = s;
}

// ---------------------------------------------------------------------------
extern "C" void kernel_launch(void* const* d_in, const int* in_sizes, int n_in,
                              void* d_out, int out_size, void* d_ws, size_t ws_size,
                              hipStream_t stream) {
    (void)in_sizes; (void)n_in; (void)out_size; (void)ws_size;

    const float* hidden = (const float*)d_in[0];   // [B,D]
    const float* enc    = (const float*)d_in[1];   // [S,B,E]
    const float* We     = (const float*)d_in[2];   // [D,E]
    const float* Wd     = (const float*)d_in[3];   // [D,D]
    const float* vvec   = (const float*)d_in[4];   // [D]

    float* ctx = (float*)d_out;                    // [B,E]   (first output)
    float* wts = (float*)d_out + BB * EE;          // [B,S]   (second output)

    // Workspace layout (~4.4 MB total).
    char* ws = (char*)d_ws;
    float*  dec    = (float*)ws;   ws += (size_t)BB * DD * sizeof(float);      // 128 KB
    __bf16* WeBf   = (__bf16*)ws;  ws += (size_t)DD * EE * sizeof(__bf16);     // 2 MB
    float*  scores = (float*)ws;   ws += (size_t)BB * SS * sizeof(float);      // 256 KB
    float*  parts  = (float*)ws;                                               // 2 MB

    bah_convert_we<<<(DD * EE) / (256 * 4), 256, 0, stream>>>(We, WeBf);
    bah_dec_proj  <<<128, 32, 0, stream>>>(hidden, Wd, dec);
    bah_scores    <<<BB * (SS / 64), 256, 0, stream>>>(enc, WeBf, dec, vvec, scores);
    bah_softmax   <<<BB, 256, 0, stream>>>(scores, wts);
    bah_ctx_partial<<<BB * 16, 256, 0, stream>>>(enc, wts, parts);
    bah_ctx_reduce <<<(BB * EE) / 256, 256, 0, stream>>>(parts, ctx);
}
